// TinyLM_19146964205732
// MI455X (gfx1250) — matrix operand order repaired and checked by reading, not compiled, not run on hardware
//
#include <hip/hip_runtime.h>
#include <hip/hip_bf16.h>

#define SEQ_L 8192
#define DIM   1024

typedef __attribute__((ext_vector_type(16))) __bf16 v16bf;
typedef __attribute__((ext_vector_type(8)))  float  v8f;
typedef unsigned short u16_t;
typedef unsigned int   u32_t;

__device__ __forceinline__ u16_t f32_to_bf16(float f) {
  unsigned u = __float_as_uint(f);
  unsigned r = u + 0x7FFFu + ((u >> 16) & 1u);   // round-to-nearest-even
  return (u16_t)(r >> 16);
}

// 32-bit wave-relative LDS byte offset of a generic pointer to __shared__ data
typedef __attribute__((address_space(3))) unsigned char* as3_ptr_t;
__device__ __forceinline__ unsigned lds_off(void* p) {
  return (unsigned)(unsigned long long)(as3_ptr_t)p;
}

// Async copy 16B global -> LDS (tracked by ASYNCcnt)
__device__ __forceinline__ void async_copy_b128(unsigned lds_byte_off, const void* gptr) {
  asm volatile("global_load_async_to_lds_b128 %0, %1, off"
               :
               : "v"(lds_byte_off), "v"(gptr)
               : "memory");
}
__device__ __forceinline__ void wait_asynccnt0() {
  asm volatile("s_wait_asynccnt 0x0" ::: "memory");
}

// ---------------- K0: convert B_w f32 -> bf16 ----------------
__global__ __launch_bounds__(256) void cvt_bw_kernel(const float* __restrict__ Bw,
                                                     u16_t* __restrict__ BwB) {
  size_t i = ((size_t)blockIdx.x * blockDim.x + threadIdx.x) * 4;
  float4 v = *(const float4*)(Bw + i);
  union { u16_t s[4]; uint2 u; } p;
  p.s[0] = f32_to_bf16(v.x); p.s[1] = f32_to_bf16(v.y);
  p.s[2] = f32_to_bf16(v.z); p.s[3] = f32_to_bf16(v.w);
  *(uint2*)(BwB + i) = p.u;
}

// ---------------- K1: embedding gather (f32 out) + bf16 copy ----------------
__global__ __launch_bounds__(256) void gather_kernel(const int* __restrict__ tokens,
                                                     const float* __restrict__ table,
                                                     float* __restrict__ embeds_out,
                                                     u16_t* __restrict__ embB) {
  int t   = blockIdx.x;
  int tok = tokens[t];
  size_t src = (size_t)tok * DIM;
  size_t dst = (size_t)t   * DIM;
  int idx = threadIdx.x * 4;
  float4 v = *(const float4*)(table + src + idx);
  *(float4*)(embeds_out + dst + idx) = v;
  union { u16_t s[4]; uint2 u; } p;
  p.s[0] = f32_to_bf16(v.x); p.s[1] = f32_to_bf16(v.y);
  p.s[2] = f32_to_bf16(v.z); p.s[3] = f32_to_bf16(v.w);
  *(uint2*)(embB + dst + idx) = p.u;
}

// ---------------- K2: bx = embeds @ B_w^T via v_wmma_f32_16x16x32_bf16 ------
// Block = 8 waves; each wave computes a 16(M) x 64(N) tile; block tile 128x64.
__global__ __launch_bounds__(256) void gemm_bx_kernel(const u16_t* __restrict__ embB,
                                                      const u16_t* __restrict__ BwB,
                                                      float* __restrict__ bx) {
  const int lane = threadIdx.x & 31;
  const int wave = threadIdx.x >> 5;
  const int hi   = lane >> 4;        // half-wave select
  const int lm   = lane & 15;
  const int i0   = blockIdx.y * 128 + wave * 16;  // M tile origin
  const int j0   = blockIdx.x * 64;               // N tile origin

  const u16_t* arow  = embB + (size_t)(i0 + lm) * DIM;
  const u16_t* brow0 = BwB  + (size_t)(j0 +  0 + lm) * DIM;
  const u16_t* brow1 = BwB  + (size_t)(j0 + 16 + lm) * DIM;
  const u16_t* brow2 = BwB  + (size_t)(j0 + 32 + lm) * DIM;
  const u16_t* brow3 = BwB  + (size_t)(j0 + 48 + lm) * DIM;

  v8f acc0 = {}, acc1 = {}, acc2 = {}, acc3 = {};
  union Frag { uint4 q[2]; v16bf v; };

  for (int k = 0; k < DIM; k += 32) {
    // A fragment (16x32 bf16): 16 contiguous K per lane, hi lanes shifted by 16
    Frag a;
    a.q[0] = *(const uint4*)(arow + k + hi * 16);
    a.q[1] = *(const uint4*)(arow + k + hi * 16 + 8);
    if (k + 64 < DIM) __builtin_prefetch(arow + k + 64, 0, 1);

    const int kb = k + hi * 16;      // B fragment: 16 contiguous K per lane
    Frag b0, b1, b2, b3;
    b0.q[0] = *(const uint4*)(brow0 + kb); b0.q[1] = *(const uint4*)(brow0 + kb + 8);
    b1.q[0] = *(const uint4*)(brow1 + kb); b1.q[1] = *(const uint4*)(brow1 + kb + 8);
    b2.q[0] = *(const uint4*)(brow2 + kb); b2.q[1] = *(const uint4*)(brow2 + kb + 8);
    b3.q[0] = *(const uint4*)(brow3 + kb); b3.q[1] = *(const uint4*)(brow3 + kb + 8);

    acc0 = __builtin_amdgcn_wmma_f32_16x16x32_bf16(false, a.v, false, b0.v, (short)0, acc0, false, false);
    acc1 = __builtin_amdgcn_wmma_f32_16x16x32_bf16(false, a.v, false, b1.v, (short)0, acc1, false, false);
    acc2 = __builtin_amdgcn_wmma_f32_16x16x32_bf16(false, a.v, false, b2.v, (short)0, acc2, false, false);
    acc3 = __builtin_amdgcn_wmma_f32_16x16x32_bf16(false, a.v, false, b3.v, (short)0, acc3, false, false);
  }

  // D layout: VGPR r -> (M = r + 8*hi, N = lm)
  #pragma unroll
  for (int r = 0; r < 8; ++r) {
    size_t row = (size_t)(i0 + r + hi * 8);
    size_t col = (size_t)(j0 + lm);
    bx[row * DIM + col +  0] = acc0[r];
    bx[row * DIM + col + 16] = acc1[r];
    bx[row * DIM + col + 32] = acc2[r];
    bx[row * DIM + col + 48] = acc3[r];
  }
}

// ---------------- K3: init barrier counter + h0 = 0 ----------------
__global__ __launch_bounds__(256) void scan_init_kernel(float* __restrict__ hbufs,
                                                        u32_t* __restrict__ ctr) {
  if (threadIdx.x == 0) *ctr = 0u;
  int idx = threadIdx.x * 4;
  *(float4*)(hbufs + idx) = make_float4(0.f, 0.f, 0.f, 0.f);   // h buffer 0
}

// ---------------- K4: persistent sequential scan -----------------------------
// 32 blocks; block b owns rows [b*32, b*32+32) of A_w, resident in LDS (f32).
#define SCAN_BLOCKS 32
#define SCAN_ROWS   32
__global__ __launch_bounds__(256) void scan_kernel(const float* __restrict__ A_w,
                                                   const float* __restrict__ bx,
                                                   float* __restrict__ hs_out,
                                                   float* __restrict__ hbufs,
                                                   u32_t* __restrict__ ctr) {
  extern __shared__ float smem[];                 // [SCAN_ROWS*DIM] A + [DIM] h
  float* As  = smem;
  float* hsh = smem + SCAN_ROWS * DIM;

  const int j0   = blockIdx.x * SCAN_ROWS;
  const int wave = threadIdx.x >> 5;
  const int lane = threadIdx.x & 31;

  const unsigned As_off  = lds_off(As);
  const unsigned hsh_off = lds_off(hsh);

  // Preload this block's A_w slice into LDS (128 KB, f32) via async-to-LDS DMA
  for (int idx = threadIdx.x * 4; idx < SCAN_ROWS * DIM; idx += blockDim.x * 4) {
    async_copy_b128(As_off + (unsigned)idx * 4u,
                    (const void*)(A_w + (size_t)j0 * DIM + idx));
  }
  wait_asynccnt0();
  __syncthreads();

  for (int t = 0; t < SEQ_L; ++t) {
    // Stage h_{t-1} (4 KB) into LDS: one async b128 per thread
    const float* hprev = hbufs + (size_t)(t & 1) * DIM;
    {
      const int idx = threadIdx.x * 4;            // 256 threads x 4 floats = DIM
      async_copy_b128(hsh_off + (unsigned)idx * 4u, (const void*)(hprev + idx));
    }
    wait_asynccnt0();
    __syncthreads();

    // 8 waves x 4 rows each: dot(A_row, h) with lane-strided accumulate
    #pragma unroll
    for (int rr = 0; rr < SCAN_ROWS / 8; ++rr) {
      const int jloc = wave * (SCAN_ROWS / 8) + rr;
      const float* arow = As + jloc * DIM;
      float acc = 0.f;
      #pragma unroll 8
      for (int k = lane; k < DIM; k += 32) acc = fmaf(arow[k], hsh[k], acc);
      #pragma unroll
      for (int off = 16; off > 0; off >>= 1) acc += __shfl_down(acc, off, 32);
      if (lane == 0) {
        const int j = j0 + jloc;
        float v = tanhf(acc + bx[(size_t)t * DIM + j]);
        hs_out[(size_t)t * DIM + j] = v;
        hbufs[(size_t)((t + 1) & 1) * DIM + j] = v;
      }
    }

    // Device-wide barrier: release our writes, spin until all 32 blocks arrive
    __threadfence();
    __syncthreads();
    if (threadIdx.x == 0) {
      __hip_atomic_fetch_add(ctr, 1u, __ATOMIC_RELEASE, __HIP_MEMORY_SCOPE_AGENT);
      const u32_t target = (u32_t)(t + 1) * SCAN_BLOCKS;
      while (__hip_atomic_load(ctr, __ATOMIC_ACQUIRE, __HIP_MEMORY_SCOPE_AGENT) < target) {
        __builtin_amdgcn_s_sleep(2);
      }
    }
    __syncthreads();   // also separates hsh reuse across iterations
  }
}

// ---------------- host-side launch ----------------
extern "C" void kernel_launch(void* const* d_in, const int* in_sizes, int n_in,
                              void* d_out, int out_size, void* d_ws, size_t ws_size,
                              hipStream_t stream) {
  const int*   tokens = (const int*)d_in[0];
  const float* table  = (const float*)d_in[1];
  const float* A_w    = (const float*)d_in[2];
  const float* B_w    = (const float*)d_in[3];

  float* hs_out     = (float*)d_out;                          // [L, D]
  float* embeds_out = (float*)d_out + (size_t)SEQ_L * DIM;    // [L, D]

  // workspace layout
  char*  ws    = (char*)d_ws;
  u16_t* embB  = (u16_t*)ws;                                              // 16 MB
  u16_t* BwB   = (u16_t*)(ws + (size_t)SEQ_L * DIM * 2);                  //  2 MB
  float* bx    = (float*)(ws + (size_t)SEQ_L * DIM * 2 + (size_t)DIM * DIM * 2); // 32 MB
  float* hbufs = (float*)((char*)bx + (size_t)SEQ_L * DIM * 4);           //  8 KB
  u32_t* ctr   = (u32_t*)((char*)hbufs + (size_t)2 * DIM * 4);

  cvt_bw_kernel<<<(DIM * DIM) / 1024, 256, 0, stream>>>(B_w, BwB);
  gather_kernel<<<SEQ_L, 256, 0, stream>>>(tokens, table, embeds_out, embB);

  dim3 ggrid(DIM / 64, SEQ_L / 128);
  gemm_bx_kernel<<<ggrid, 256, 0, stream>>>(embB, BwB, bx);

  scan_init_kernel<<<1, 256, 0, stream>>>(hbufs, ctr);

  size_t lds_bytes = (size_t)(SCAN_ROWS * DIM + DIM) * sizeof(float);     // 132 KB
  scan_kernel<<<SCAN_BLOCKS, 256, lds_bytes, stream>>>(A_w, bx, hs_out, hbufs, ctr);
}
